// MultiHeadAttention_23897198035557
// MI455X (gfx1250) — matrix operand order repaired and checked by reading, not compiled, not run on hardware
//
#include <hip/hip_runtime.h>

#define DEV __device__ __forceinline__

typedef __bf16 bf16;
typedef __attribute__((ext_vector_type(16))) __bf16 v16bf;
typedef __attribute__((ext_vector_type(8)))  float  v8f;
typedef int gi4 __attribute__((vector_size(16)));   // GNU int4, matches builtin proto

#define AS1 __attribute__((address_space(1)))
#define AS3 __attribute__((address_space(3)))

namespace {
constexpr int Bc = 4, Sc = 2048, Dc = 1024, Hc = 16, DHc = 64;
constexpr int Mrows = Bc * Sc;  // 8192
}

#if defined(__AMDGCN__) && __has_builtin(__builtin_amdgcn_global_load_async_to_lds_b128)
#define ASYNC_COPY 1
#else
#define ASYNC_COPY 0
#endif

// ---------- helpers ----------
DEV bf16 f2bf(float f) {
  unsigned u = __builtin_bit_cast(unsigned, f);
  unsigned r = u + 0x7FFFu + ((u >> 16) & 1u);   // round-to-nearest-even
  unsigned short h = (unsigned short)(r >> 16);
  return __builtin_bit_cast(bf16, h);
}

DEV v8f wmma_bf16(v16bf a, v16bf b, v8f c) {
  // D = A(16x32 bf16) x B(32x16 bf16) + C(16x16 f32)
  return __builtin_amdgcn_wmma_f32_16x16x32_bf16(
      false, a, false, b, (short)0, c, false, false);
}

DEV void store_out(float* p, float v) { *p = v; }
DEV void store_out(bf16*  p, float v) { *p = f2bf(v); }

// 16-byte copy global -> LDS. Async (ASYNCcnt-tracked, no VGPR round trip) when
// the gfx1250 builtin exists; otherwise a register-staged copy.
DEV void cp_b128(void* l, const void* g) {
#if ASYNC_COPY
  __builtin_amdgcn_global_load_async_to_lds_b128(
      (AS1 gi4*)(unsigned long long)g,
      (AS3 gi4*)(unsigned)(unsigned long long)l,
      0, 0);
#else
  *(uint4*)l = *(const uint4*)g;
#endif
}
DEV void cp_commit_wait() {
#if ASYNC_COPY
#if __has_builtin(__builtin_amdgcn_s_wait_asynccnt)
  __builtin_amdgcn_s_wait_asynccnt(0);
#else
  asm volatile("s_wait_asynccnt 0x0" ::: "memory");
#endif
#endif
}

// A fragment (16x32, MxK): lane holds row M=lane&15; K = kh*8+{0..7} and kh*8+16+{0..7}
DEV v16bf lds_frag_a(const bf16* rowbase, int lane, int stride) {
  const bf16* p = rowbase + (lane & 15) * stride + (lane >> 4) * 8;
  union { v16bf v; uint4 q[2]; } u;
  u.q[0] = *(const uint4*)p;
  u.q[1] = *(const uint4*)(p + 16);
  return u.v;
}
// B fragment (32x16, KxN) from K-contiguous rows (B^T storage): lane holds
// column N=lane&15; K = kh*8+{0..7} and kh*8+16+{0..7}.
DEV v16bf lds_frag_b(const bf16* colbase, int lane, int stride) {
  const bf16* p = colbase + (lane & 15) * stride + (lane >> 4) * 8;
  union { v16bf v; uint4 q[2]; } u;
  u.q[0] = *(const uint4*)p;
  u.q[1] = *(const uint4*)(p + 16);
  return u.v;
}

// ---------- elementwise fp32 -> bf16 ----------
__global__ __launch_bounds__(256) void k_cvt_bf16(const float* __restrict__ in,
                                                  bf16* __restrict__ out, int n) {
  int i = blockIdx.x * 256 + threadIdx.x;
  int stride = gridDim.x * 256;
  for (; i < n; i += stride) out[i] = f2bf(in[i]);
}

// ---------- convert + transpose weight: WT[n][k] = W[k][n] (dim x dim) ----------
__global__ __launch_bounds__(256) void k_cvt_transpose(const float* __restrict__ W,
                                                       bf16* __restrict__ WT, int dim) {
  __shared__ bf16 tile[64][72];
  const int k0 = blockIdx.y * 64, n0 = blockIdx.x * 64;
  const int tid = threadIdx.x;
#pragma unroll
  for (int i = 0; i < 16; ++i) {
    int e = i * 256 + tid;
    int kl = e >> 6, nl = e & 63;
    tile[kl][nl] = f2bf(W[(size_t)(k0 + kl) * dim + n0 + nl]);
  }
  __syncthreads();
#pragma unroll
  for (int i = 0; i < 16; ++i) {
    int e = i * 256 + tid;
    int nl = e >> 6, kl = e & 63;
    WT[(size_t)(n0 + nl) * dim + k0 + kl] = tile[kl][nl];
  }
}

// ---------- per-head V transpose: Vt[b,h,d,s] = V[b,s,h*64+d] ----------
__global__ __launch_bounds__(256) void k_transpose_v(const bf16* __restrict__ V,
                                                     bf16* __restrict__ Vt) {
  __shared__ bf16 tile[64][72];
  const int bh = blockIdx.y, b = bh >> 4, h = bh & 15;
  const int s0 = blockIdx.x * 64;
  const int tid = threadIdx.x;
#pragma unroll
  for (int i = 0; i < 16; ++i) {
    int e = i * 256 + tid;
    int sl = e >> 6, dl = e & 63;
    tile[sl][dl] = V[((size_t)(b * Sc + s0 + sl)) * Dc + h * DHc + dl];
  }
  __syncthreads();
#pragma unroll
  for (int i = 0; i < 16; ++i) {
    int e = i * 256 + tid;
    int dl = e >> 6, sl = e & 63;
    Vt[((size_t)((b * Hc + h) * DHc + dl)) * Sc + s0 + sl] = tile[sl][dl];
  }
}

// ---------- GEMM: C[M,N] = A[M,K](bf16) * BT[N,K]^T (bf16) + bias[N] ----------
// Block tile 128x128, 8 waves (4m x 2n), wave tile 32x64, K-step 32.
// Double-buffered LDS staging via async global->LDS copies.
template <typename OutT>
__global__ __launch_bounds__(256) void k_gemm_bias(
    const bf16* __restrict__ A, const bf16* __restrict__ BT,
    const float* __restrict__ bias, OutT* __restrict__ C,
    int M, int N, int K) {
  constexpr int BM = 128, BN = 128, BK = 32, LST = 40;  // padded LDS row stride
  __shared__ bf16 As[2][BM * LST];
  __shared__ bf16 Bs[2][BN * LST];
  const int tid = threadIdx.x, lane = tid & 31, wid = tid >> 5;
  const int wm = wid >> 1, wn = wid & 1;
  const int mblock = blockIdx.y * BM, nblock = blockIdx.x * BN;
  const int lrow = tid >> 1, lhalf = tid & 1;   // 2 threads stage one 64B LDS row

  const bf16* gA = A + (size_t)(mblock + lrow) * K + lhalf * 16;
  const bf16* gB = BT + (size_t)(nblock + lrow) * K + lhalf * 16;

  auto stage = [&](int k0, int bsel) {
    const bf16* ga = gA + k0;
    const bf16* gb = gB + k0;
    bf16* la = &As[bsel][lrow * LST + lhalf * 16];
    bf16* lb = &Bs[bsel][lrow * LST + lhalf * 16];
    cp_b128(la,     ga);
    cp_b128(la + 8, ga + 8);
    cp_b128(lb,     gb);
    cp_b128(lb + 8, gb + 8);
  };

  v8f acc[2][4];
  {
    v8f z = {};
#pragma unroll
    for (int i = 0; i < 2; ++i)
#pragma unroll
      for (int j = 0; j < 4; ++j) acc[i][j] = z;
  }

  stage(0, 0);
  cp_commit_wait();
  __syncthreads();

  int buf = 0;
  for (int k0 = 0; k0 < K; k0 += BK) {
    if (k0 + BK < K) stage(k0 + BK, buf ^ 1);

    v16bf af[2], bfv[4];
#pragma unroll
    for (int fm = 0; fm < 2; ++fm)
      af[fm] = lds_frag_a(&As[buf][(wm * 32 + fm * 16) * LST], lane, LST);
#pragma unroll
    for (int fn = 0; fn < 4; ++fn)
      bfv[fn] = lds_frag_b(&Bs[buf][(wn * 64 + fn * 16) * LST], lane, LST);
#pragma unroll
    for (int fm = 0; fm < 2; ++fm)
#pragma unroll
      for (int fn = 0; fn < 4; ++fn)
        acc[fm][fn] = wmma_bf16(af[fm], bfv[fn], acc[fm][fn]);

    cp_commit_wait();
    __syncthreads();
    buf ^= 1;
  }

#pragma unroll
  for (int fm = 0; fm < 2; ++fm) {
#pragma unroll
    for (int fn = 0; fn < 4; ++fn) {
      const int n = nblock + wn * 64 + fn * 16 + (lane & 15);
      const float bv = bias[n];
      const int mbase = mblock + wm * 32 + fm * 16 + (lane >> 4) * 8;
#pragma unroll
      for (int r = 0; r < 8; ++r)
        store_out(&C[(size_t)(mbase + r) * N + n], acc[fm][fn][r] + bv);
    }
  }
}

// ---------- flash attention: per (b,h), 128 queries/block, 16 per wave ----------
// K/V tiles staged once per block into double-buffered LDS (async), P tile
// bridged through per-wave LDS from WMMA C-layout to A-layout.
__global__ __launch_bounds__(256) void k_flash_attn(
    const bf16* __restrict__ K, const bf16* __restrict__ Q,
    const bf16* __restrict__ Vt, bf16* __restrict__ Ctx) {
  constexpr float scale = 0.125f;  // 1/sqrt(64)
  constexpr int ST = 72;           // padded LDS stride (elements)
  __shared__ bf16 Ks[2][64 * ST];  // [key 0..63][dh 0..63]
  __shared__ bf16 Vs[2][64 * ST];  // [dh 0..63][key 0..63]
  __shared__ bf16 Plds[8][16 * 80];
  const int tid = threadIdx.x;
  const int lane = tid & 31, wid = tid >> 5;
  const int bh = blockIdx.y, b = bh >> 4, h = bh & 15;
  const int q0 = blockIdx.x * 128 + wid * 16;
  const int cl = lane & 15, kh = lane >> 4;
  const int srow = tid >> 2, spart = tid & 3;  // staging: 64 rows x 4x32B

  const bf16* Kbase = K + ((size_t)(b * Sc)) * Dc + h * DHc;
  const bf16* Vbase = Vt + ((size_t)((b * Hc + h) * DHc + srow)) * Sc;

  auto stage = [&](int j0, int bsel) {
    const bf16* gk = Kbase + (size_t)(j0 + srow) * Dc + spart * 16;
    const bf16* gv = Vbase + j0 + spart * 16;
    bf16* lk = &Ks[bsel][srow * ST + spart * 16];
    bf16* lv = &Vs[bsel][srow * ST + spart * 16];
    cp_b128(lk,     gk);
    cp_b128(lk + 8, gk + 8);
    cp_b128(lv,     gv);
    cp_b128(lv + 8, gv + 8);
  };

  // Q A-fragments for this wave's 16 queries (DH=64 -> two K=32 slices)
  v16bf qf[2];
  {
    const bf16* qp = Q + ((size_t)(b * Sc + q0 + cl)) * Dc + h * DHc + kh * 8;
#pragma unroll
    for (int ks = 0; ks < 2; ++ks) {
      union { v16bf v; uint4 u[2]; } t;
      t.u[0] = *(const uint4*)(qp + ks * 32);
      t.u[1] = *(const uint4*)(qp + ks * 32 + 16);
      qf[ks] = t.v;
    }
  }

  v8f cacc[4];
  {
    v8f z = {};
#pragma unroll
    for (int fd = 0; fd < 4; ++fd) cacc[fd] = z;
  }
  float mrow[8], lsum[8];
#pragma unroll
  for (int r = 0; r < 8; ++r) { mrow[r] = -1.0e30f; lsum[r] = 0.0f; }

  stage(0, 0);
  cp_commit_wait();
  __syncthreads();

  int buf = 0;
  for (int j0 = 0; j0 < Sc; j0 += 64) {
    if (j0 + 64 < Sc) stage(j0 + 64, buf ^ 1);

    // ---- scores S = Q K^T (16 x 64) from LDS K tile ----
    v8f sc[4];
    {
      v8f z = {};
#pragma unroll
      for (int fn = 0; fn < 4; ++fn) sc[fn] = z;
    }
#pragma unroll
    for (int ks = 0; ks < 2; ++ks) {
#pragma unroll
      for (int fn = 0; fn < 4; ++fn) {
        const bf16* kp = &Ks[buf][(fn * 16 + cl) * ST + ks * 32 + kh * 8];
        union { v16bf v; uint4 u[2]; } t;
        t.u[0] = *(const uint4*)kp;
        t.u[1] = *(const uint4*)(kp + 16);
        sc[fn] = wmma_bf16(qf[ks], t.v, sc[fn]);
      }
    }

    // ---- online softmax (row lives across a 16-lane group) ----
    float alpha[8];
#pragma unroll
    for (int r = 0; r < 8; ++r) {
      float t = fmaxf(fmaxf(sc[0][r], sc[1][r]), fmaxf(sc[2][r], sc[3][r])) * scale;
      t = fmaxf(t, __shfl_xor(t, 1));
      t = fmaxf(t, __shfl_xor(t, 2));
      t = fmaxf(t, __shfl_xor(t, 4));
      t = fmaxf(t, __shfl_xor(t, 8));
      const float mn = fmaxf(mrow[r], t);
      alpha[r] = __expf(mrow[r] - mn);
      mrow[r] = mn;
    }
    float rsum[8] = {0, 0, 0, 0, 0, 0, 0, 0};
    bf16* prow = &Plds[wid][0];
#pragma unroll
    for (int fn = 0; fn < 4; ++fn) {
#pragma unroll
      for (int r = 0; r < 8; ++r) {
        const float pv = __expf(sc[fn][r] * scale - mrow[r]);
        rsum[r] += pv;
        prow[(r + 8 * kh) * 80 + fn * 16 + cl] = f2bf(pv);  // C-layout -> LDS
      }
    }
#pragma unroll
    for (int r = 0; r < 8; ++r) {
      float s = rsum[r];
      s += __shfl_xor(s, 1);
      s += __shfl_xor(s, 2);
      s += __shfl_xor(s, 4);
      s += __shfl_xor(s, 8);
      lsum[r] = lsum[r] * alpha[r] + s;
    }
#pragma unroll
    for (int fd = 0; fd < 4; ++fd)
#pragma unroll
      for (int r = 0; r < 8; ++r) cacc[fd][r] *= alpha[r];

    // ---- ctx += P (16x64) x V (64x64); P tile is wave-private, DS ops from one
    // wave complete in order, so no block barrier is needed between store/load ----
#pragma unroll
    for (int ks = 0; ks < 2; ++ks) {
      union { v16bf v; uint4 u[2]; } pa;
      const bf16* pp = &Plds[wid][cl * 80 + ks * 32 + kh * 8];
      pa.u[0] = *(const uint4*)pp;
      pa.u[1] = *(const uint4*)(pp + 16);
#pragma unroll
      for (int fd = 0; fd < 4; ++fd) {
        const bf16* vp = &Vs[buf][(fd * 16 + cl) * ST + ks * 32 + kh * 8];
        union { v16bf v; uint4 u[2]; } vb2;
        vb2.u[0] = *(const uint4*)vp;
        vb2.u[1] = *(const uint4*)(vp + 16);
        cacc[fd] = wmma_bf16(pa.v, vb2.v, cacc[fd]);
      }
    }

    cp_commit_wait();
    __syncthreads();
    buf ^= 1;
  }

  // ---- normalize and write ctx (bf16, [b,s,h*64+d]) ----
#pragma unroll
  for (int r = 0; r < 8; ++r) {
    const float inv = 1.0f / lsum[r];
    const int row = q0 + r + 8 * kh;
#pragma unroll
    for (int fd = 0; fd < 4; ++fd)
      Ctx[((size_t)(b * Sc + row)) * Dc + h * DHc + fd * 16 + cl] =
          f2bf(cacc[fd][r] * inv);
  }
}

// ---------- host launcher ----------
extern "C" void kernel_launch(void* const* d_in, const int* in_sizes, int n_in,
                              void* d_out, int out_size, void* d_ws, size_t ws_size,
                              hipStream_t stream) {
  (void)in_sizes; (void)n_in; (void)out_size; (void)ws_size;
  const float* x  = (const float*)d_in[0];
  const float* Wq = (const float*)d_in[1];
  const float* bq = (const float*)d_in[2];
  const float* Wk = (const float*)d_in[3];
  const float* bk = (const float*)d_in[4];
  const float* Wv = (const float*)d_in[5];
  const float* bv = (const float*)d_in[6];
  const float* Wo = (const float*)d_in[7];
  const float* bo = (const float*)d_in[8];
  float* out = (float*)d_out;

  const size_t nX = (size_t)Bc * Sc * Dc;  // 8M elems
  const size_t nW = (size_t)Dc * Dc;       // 1M elems
  char* p = (char*)d_ws;
  bf16* xbf = (bf16*)p; p += nX * 2;
  bf16* WqT = (bf16*)p; p += nW * 2;
  bf16* WkT = (bf16*)p; p += nW * 2;
  bf16* WvT = (bf16*)p; p += nW * 2;
  bf16* WoT = (bf16*)p; p += nW * 2;
  bf16* Qb  = (bf16*)p; p += nX * 2;
  bf16* Kb  = (bf16*)p; p += nX * 2;
  bf16* Vb  = (bf16*)p; p += nX * 2;
  bf16* Vtb = (bf16*)p; p += nX * 2;
  bf16* Cxb = (bf16*)p; p += nX * 2;

  k_cvt_bf16<<<4096, 256, 0, stream>>>(x, xbf, (int)nX);
  dim3 tw(Dc / 64, Dc / 64);
  k_cvt_transpose<<<tw, 256, 0, stream>>>(Wq, WqT, Dc);
  k_cvt_transpose<<<tw, 256, 0, stream>>>(Wk, WkT, Dc);
  k_cvt_transpose<<<tw, 256, 0, stream>>>(Wv, WvT, Dc);
  k_cvt_transpose<<<tw, 256, 0, stream>>>(Wo, WoT, Dc);

  dim3 gg(Dc / 128, Mrows / 128);  // (8, 64)
  k_gemm_bias<bf16><<<gg, 256, 0, stream>>>(xbf, WqT, bq, Qb, Mrows, Dc, Dc);
  k_gemm_bias<bf16><<<gg, 256, 0, stream>>>(xbf, WkT, bk, Kb, Mrows, Dc, Dc);
  k_gemm_bias<bf16><<<gg, 256, 0, stream>>>(xbf, WvT, bv, Vb, Mrows, Dc, Dc);

  k_transpose_v<<<dim3(Sc / 64, Bc * Hc), 256, 0, stream>>>(Vb, Vtb);
  k_flash_attn<<<dim3(Sc / 128, Bc * Hc), 256, 0, stream>>>(Kb, Qb, Vtb, Cxb);

  k_gemm_bias<float><<<gg, 256, 0, stream>>>(Cxb, WoT, bo, out, Mrows, Dc, Dc);
}
